// MoELayer_11003706212941
// MI455X (gfx1250) — matrix operand order repaired and checked by reading, not compiled, not run on hardware
//
#include <hip/hip_runtime.h>
#include <hip/hip_bf16.h>
#include <stdint.h>

#define HID   1024
#define DFF   4096
#define NEXP  8
#define NTOK  4096
#define TM    16          // tokens per tile
#define KT    32          // K chunk per stage
#define NH    2048        // half of DFF held in LDS
#define NHP   2052        // padded row stride for Hs (bank spread)
#define TPB   256
#define SBCH  (KT * 1024) // elements per staging buffer

typedef __attribute__((ext_vector_type(16))) __bf16 v16bf;
typedef __attribute__((ext_vector_type(8)))  float  v8f;

// fp32 -> bf16 round-to-nearest-even (bit ops)
__device__ __forceinline__ unsigned short f32_to_bf16(float f) {
  unsigned int u = __builtin_bit_cast(unsigned int, f);
  u = u + 0x7FFFu + ((u >> 16) & 1u);
  return (unsigned short)(u >> 16);
}

// A fragment: 16x32 bf16, rows m=0..15 row-major in LDS with 'stride' (ushort units).
__device__ __forceinline__ v16bf frag_a(const unsigned short* base, int stride,
                                        int k0, int lane) {
  union { v16bf v; unsigned int u[8]; } f;
  const int m  = lane & 15;
  const int hi = (lane >> 4) & 1;
  const unsigned short* p = base + m * stride + k0 + 8 * hi;
  #pragma unroll
  for (int r = 0; r < 8; ++r) {
    const int kb = 2 * (r & 3) + 16 * (r >> 2);
    f.u[r] = *(const unsigned int*)(p + kb);
  }
  return f.v;
}

// B fragment: 32x16 bf16 from LDS staged as [n][k] (row stride KT).
__device__ __forceinline__ v16bf frag_b(const unsigned short* sb, int ncol0, int lane) {
  union { v16bf v; unsigned int u[8]; } f;
  const int n  = ncol0 + (lane & 15);
  const int hi = (lane >> 4) & 1;
  const unsigned short* p = sb + n * KT + 8 * hi;
  #pragma unroll
  for (int r = 0; r < 8; ++r) f.u[r] = *(const unsigned int*)(p + 2 * (r & 3) + 16 * (r >> 2));
  return f.v;
}

// Async global->LDS copy, 16B per lane, GVS mode (SGPR base + VGPR offset).
__device__ __forceinline__ void async_load_b128(unsigned int lds_off,
                                                const void* sbase,
                                                unsigned int voff) {
  asm volatile("global_load_async_to_lds_b128 %0, %1, %2"
               :: "v"(lds_off), "v"(voff), "s"(sbase) : "memory");
}
__device__ __forceinline__ void wait_asynccnt0() {
  asm volatile("s_wait_asynccnt 0x0" ::: "memory");
}

__global__ void moe_zero_kernel(float* __restrict__ out, int n, int* __restrict__ counts) {
  const int i = blockIdx.x * blockDim.x + threadIdx.x;
  if (i < n) out[i] = 0.0f;
  if (i < NEXP) counts[i] = 0;
}

__global__ void moe_route_kernel(const float* __restrict__ x,
                                 int* __restrict__ counts, int* __restrict__ list) {
  const int t = blockIdx.x * blockDim.x + threadIdx.x;
  if (t >= NTOK) return;
  const float s = x[(size_t)t * HID + 0] + x[(size_t)t * HID + 1];
  int h = (int)s;                    // trunc toward zero, like astype(int32)
  h = ((h % NEXP) + NEXP) % NEXP;    // floor-mod, like jnp.mod
  const int e0 = h, e1 = (h + 1) % NEXP;
  int p0 = atomicAdd(&counts[e0], 1); list[e0 * NTOK + p0] = t;
  int p1 = atomicAdd(&counts[e1], 1); list[e1 * NTOK + p1] = t;
}

// Tiled transpose + fp32->bf16: src [R][C] fp32 -> dst [C][R] bf16, per expert (grid.z)
__global__ void transpose_bf16_kernel(const float* __restrict__ src,
                                      unsigned short* __restrict__ dst,
                                      int R, int C) {
  __shared__ unsigned short tile[32][33];
  const int e = blockIdx.z;
  src += (size_t)e * R * C;
  dst += (size_t)e * R * C;
  const int bx = blockIdx.x * 32;        // col base in src
  const int by = blockIdx.y * 32;        // row base in src
  const int tx = threadIdx.x & 31;
  const int ty = threadIdx.x >> 5;       // 0..7
  #pragma unroll
  for (int s = 0; s < 32; s += 8)
    tile[ty + s][tx] = f32_to_bf16(src[(size_t)(by + ty + s) * C + (bx + tx)]);
  __syncthreads();
  #pragma unroll
  for (int s = 0; s < 32; s += 8)
    dst[(size_t)(bx + ty + s) * R + (by + tx)] = tile[tx][ty + s];
}

// ---------------- Fast path: bf16 pre-transposed weights + double-buffered async staging
__global__ __launch_bounds__(TPB)
void moe_ffn_async_kernel(const float* __restrict__ x,
                          const unsigned short* __restrict__ W1t, // [E][DFF][HID] bf16
                          const float* __restrict__ b1,
                          const unsigned short* __restrict__ W2t, // [E][HID][DFF] bf16
                          const float* __restrict__ b2,
                          float* __restrict__ out,
                          const int* __restrict__ counts,
                          const int* __restrict__ list) {
  __shared__ unsigned short Xs[TM * HID];    // 32 KB
  __shared__ unsigned short Hs[TM * NHP];    // ~64 KB
  __shared__ unsigned short SB[2][SBCH];     // 2 x 64 KB, [n][k]
  __shared__ int toks[TM];

  const int e    = blockIdx.x >> 8;
  const int tile = blockIdx.x & 255;
  const int nTok = counts[e];
  if (tile * TM >= nTok) return;

  const int tid  = threadIdx.x;
  const int lane = tid & 31;
  const int wave = tid >> 5;
  const int hi   = (lane >> 4) & 1;
  const int lm   = lane & 15;

  const unsigned short* W1te = W1t + (size_t)e * DFF * HID;
  const unsigned short* W2te = W2t + (size_t)e * HID * DFF;
  const float* b1e = b1 + (size_t)e * DFF;
  const float* b2e = b2 + (size_t)e * HID;

  if (tid < TM) {
    const int idx = tile * TM + tid;
    toks[tid] = (idx < nTok) ? list[e * NTOK + idx] : -1;
  }
  __syncthreads();

  for (int i = tid; i < TM * HID; i += TPB) {
    const int m = i >> 10, k = i & (HID - 1);
    const int t = toks[m];
    const float v = (t >= 0) ? x[(size_t)t * HID + k] : 0.0f;
    Xs[m * HID + k] = f32_to_bf16(v);
  }

  const unsigned int sb0 = (unsigned int)(size_t)&SB[0][0];
  const unsigned int q16 = (unsigned)(tid & 3) * 16u;
  const int nl0 = tid >> 2;

  // Issue 16 fire-and-forget b128 copies: 1024 rows x 64B into SB[buf]
  auto stage = [&](int buf, const unsigned short* gb, int rs) {
    #pragma unroll
    for (int it = 0; it < 16; ++it) {
      const int nl = nl0 + it * 64;
      async_load_b128(sb0 + (unsigned)(buf * (SBCH * 2) + nl * (KT * 2)) + q16,
                      gb, (unsigned)(nl * (rs * 2)) + q16);
    }
  };

  v8f acc2[8];
  #pragma unroll
  for (int nt = 0; nt < 8; ++nt)
    #pragma unroll
    for (int r = 0; r < 8; ++r) acc2[nt][r] = 0.0f;

  for (int half = 0; half < 2; ++half) {
    const int nOff = half * NH;

    // ---- Phase 1: Hs = relu(X @ W1[:, nOff:nOff+NH] + b1) ----
    for (int nblk = 0; nblk < NH; nblk += 1024) {
      v8f acc1[8];
      #pragma unroll
      for (int nt = 0; nt < 8; ++nt)
        #pragma unroll
        for (int r = 0; r < 8; ++r) acc1[nt][r] = 0.0f;

      const unsigned short* w1base = W1te + (size_t)(nOff + nblk) * HID;
      __syncthreads();                       // prior readers of SB[0] done
      stage(0, w1base, HID);                 // prologue: chunk 0
      int cur = 0;
      for (int k0 = 0; k0 < HID; k0 += KT) {
        wait_asynccnt0();                    // my fills of SB[cur] complete
        __syncthreads();                     // everyone's fills complete
        if (k0 + KT < HID)
          stage(cur ^ 1, w1base + (k0 + KT), HID);   // overlap next chunk

        const v16bf a = frag_a(Xs, HID, k0, lane);
        const unsigned short* sbc = &SB[cur][0];
        #pragma unroll
        for (int nt = 0; nt < 8; ++nt) {
          const v16bf bm = frag_b(sbc, wave * 128 + nt * 16, lane);
          acc1[nt] = __builtin_amdgcn_wmma_f32_16x16x32_bf16(
              false, a, false, bm, (short)0, acc1[nt], false, false);
        }
        cur ^= 1;
      }

      #pragma unroll
      for (int nt = 0; nt < 8; ++nt) {
        const int nloc = nblk + wave * 128 + nt * 16 + lm;
        const float bb = b1e[nOff + nloc];
        #pragma unroll
        for (int r = 0; r < 8; ++r) {
          float h = acc1[nt][r] + bb;
          h = h > 0.0f ? h : 0.0f;
          Hs[(r + 8 * hi) * NHP + nloc] = f32_to_bf16(h);
        }
      }
    }

    // ---- Phase 2 (partial K): acc2 += H @ W2[nOff:nOff+NH, :] ----
    {
      __syncthreads();                       // Hs writes visible, SB[0] readers done
      stage(0, W2te + (size_t)nOff, DFF);    // prologue: chunk 0
      int cur = 0;
      for (int k0 = 0; k0 < NH; k0 += KT) {
        wait_asynccnt0();
        __syncthreads();
        if (k0 + KT < NH)
          stage(cur ^ 1, W2te + (size_t)(nOff + k0 + KT), DFF);

        const v16bf a = frag_a(Hs, NHP, k0, lane);
        const unsigned short* sbc = &SB[cur][0];
        #pragma unroll
        for (int nt = 0; nt < 8; ++nt) {
          const v16bf bm = frag_b(sbc, wave * 128 + nt * 16, lane);
          acc2[nt] = __builtin_amdgcn_wmma_f32_16x16x32_bf16(
              false, a, false, bm, (short)0, acc2[nt], false, false);
        }
        cur ^= 1;
      }
    }
  }

  #pragma unroll
  for (int nt = 0; nt < 8; ++nt) {
    const int n = wave * 128 + nt * 16 + lm;
    const float bb = b2e[n];
    #pragma unroll
    for (int r = 0; r < 8; ++r) {
      const int row = r + 8 * hi;
      const int t = toks[row];
      if (t >= 0) {
        const float y = 0.5f * (acc2[nt][r] + bb);
        atomicAdd(&out[(size_t)t * HID + n], y);
      }
    }
  }
}

// ---------------- Fallback: on-the-fly fp32->bf16 staging ----
__global__ __launch_bounds__(TPB)
void moe_ffn_kernel(const float* __restrict__ x,
                    const float* __restrict__ W1,
                    const float* __restrict__ b1,
                    const float* __restrict__ W2,
                    const float* __restrict__ b2,
                    float* __restrict__ out,
                    const int* __restrict__ counts,
                    const int* __restrict__ list) {
  __shared__ unsigned short Xs[TM * HID];
  __shared__ unsigned short Hs[TM * NHP];
  __shared__ unsigned short SBf[SBCH];
  __shared__ int toks[TM];

  const int e    = blockIdx.x >> 8;
  const int tile = blockIdx.x & 255;
  const int nTok = counts[e];
  if (tile * TM >= nTok) return;

  const int tid  = threadIdx.x;
  const int lane = tid & 31;
  const int wave = tid >> 5;
  const int hi   = (lane >> 4) & 1;
  const int lm   = lane & 15;

  const float* W1e = W1 + (size_t)e * HID * DFF;
  const float* W2e = W2 + (size_t)e * DFF * HID;
  const float* b1e = b1 + (size_t)e * DFF;
  const float* b2e = b2 + (size_t)e * HID;

  if (tid < TM) {
    const int idx = tile * TM + tid;
    toks[tid] = (idx < nTok) ? list[e * NTOK + idx] : -1;
  }
  __syncthreads();

  for (int i = tid; i < TM * HID; i += TPB) {
    const int m = i >> 10, k = i & (HID - 1);
    const int t = toks[m];
    const float v = (t >= 0) ? x[(size_t)t * HID + k] : 0.0f;
    Xs[m * HID + k] = f32_to_bf16(v);
  }

  v8f acc2[8];
  #pragma unroll
  for (int nt = 0; nt < 8; ++nt)
    #pragma unroll
    for (int r = 0; r < 8; ++r) acc2[nt][r] = 0.0f;

  for (int half = 0; half < 2; ++half) {
    const int nOff = half * NH;

    for (int nblk = 0; nblk < NH; nblk += 1024) {
      v8f acc1[8];
      #pragma unroll
      for (int nt = 0; nt < 8; ++nt)
        #pragma unroll
        for (int r = 0; r < 8; ++r) acc1[nt][r] = 0.0f;

      for (int k0 = 0; k0 < HID; k0 += KT) {
        __syncthreads();
        const float* wsrc = W1e + (size_t)k0 * DFF + nOff + nblk;
        for (int i = tid; i < KT * 1024; i += TPB) {
          const int kk = i >> 10, nl = i & 1023;
          SBf[nl * KT + kk] = f32_to_bf16(wsrc[(size_t)kk * DFF + nl]);
        }
        if (k0 + KT < HID)
          __builtin_prefetch(W1e + (size_t)(k0 + KT) * DFF + nOff + nblk + tid * 4, 0, 0);
        __syncthreads();

        const v16bf a = frag_a(Xs, HID, k0, lane);
        #pragma unroll
        for (int nt = 0; nt < 8; ++nt) {
          const v16bf bm = frag_b(SBf, wave * 128 + nt * 16, lane);
          acc1[nt] = __builtin_amdgcn_wmma_f32_16x16x32_bf16(
              false, a, false, bm, (short)0, acc1[nt], false, false);
        }
      }

      #pragma unroll
      for (int nt = 0; nt < 8; ++nt) {
        const int nloc = nblk + wave * 128 + nt * 16 + lm;
        const float bb = b1e[nOff + nloc];
        #pragma unroll
        for (int r = 0; r < 8; ++r) {
          float h = acc1[nt][r] + bb;
          h = h > 0.0f ? h : 0.0f;
          Hs[(r + 8 * hi) * NHP + nloc] = f32_to_bf16(h);
        }
      }
    }

    for (int k0 = 0; k0 < NH; k0 += KT) {
      __syncthreads();
      const float* wsrc = W2e + (size_t)(nOff + k0) * HID;
      for (int i = tid; i < KT * 1024; i += TPB) {
        const int kk = i >> 10, nl = i & 1023;
        SBf[nl * KT + kk] = f32_to_bf16(wsrc[(size_t)kk * HID + nl]);
      }
      if (nOff + k0 + KT < DFF)
        __builtin_prefetch(W2e + (size_t)(nOff + k0 + KT) * HID + tid * 4, 0, 0);
      __syncthreads();

      const v16bf a = frag_a(Hs, NHP, k0, lane);
      #pragma unroll
      for (int nt = 0; nt < 8; ++nt) {
        const v16bf bm = frag_b(SBf, wave * 128 + nt * 16, lane);
        acc2[nt] = __builtin_amdgcn_wmma_f32_16x16x32_bf16(
            false, a, false, bm, (short)0, acc2[nt], false, false);
      }
    }
  }

  #pragma unroll
  for (int nt = 0; nt < 8; ++nt) {
    const int n = wave * 128 + nt * 16 + lm;
    const float bb = b2e[n];
    #pragma unroll
    for (int r = 0; r < 8; ++r) {
      const int row = r + 8 * hi;
      const int t = toks[row];
      if (t >= 0) {
        const float y = 0.5f * (acc2[nt][r] + bb);
        atomicAdd(&out[(size_t)t * HID + n], y);
      }
    }
  }
}

extern "C" void kernel_launch(void* const* d_in, const int* in_sizes, int n_in,
                              void* d_out, int out_size, void* d_ws, size_t ws_size,
                              hipStream_t stream) {
  const float* x  = (const float*)d_in[0];
  const float* W1 = (const float*)d_in[1];
  const float* b1 = (const float*)d_in[2];
  const float* W2 = (const float*)d_in[3];
  const float* b2 = (const float*)d_in[4];
  float* out = (float*)d_out;

  int* counts = (int*)d_ws;                               // 8 ints
  int* list   = (int*)((char*)d_ws + 256);                // 8*4096 ints
  const size_t wOff = 256 + (size_t)NEXP * NTOK * 4;      // 131328, 16B aligned
  unsigned short* W1t = (unsigned short*)((char*)d_ws + wOff);
  unsigned short* W2t = W1t + (size_t)NEXP * HID * DFF;
  const size_t need = wOff + (size_t)2 * NEXP * HID * DFF * sizeof(unsigned short);

  const int zblocks = (out_size + TPB - 1) / TPB;
  moe_zero_kernel<<<zblocks, TPB, 0, stream>>>(out, out_size, counts);
  moe_route_kernel<<<NTOK / TPB, TPB, 0, stream>>>(x, counts, list);

  if (ws_size >= need) {
    dim3 g1(DFF / 32, HID / 32, NEXP);   // W1 [HID][DFF] -> W1t [DFF][HID]
    transpose_bf16_kernel<<<g1, TPB, 0, stream>>>(W1, W1t, HID, DFF);
    dim3 g2(HID / 32, DFF / 32, NEXP);   // W2 [DFF][HID] -> W2t [HID][DFF]
    transpose_bf16_kernel<<<g2, TPB, 0, stream>>>(W2, W2t, DFF, HID);
    moe_ffn_async_kernel<<<NEXP * 256, TPB, 0, stream>>>(x, W1t, b1, W2t, b2,
                                                         out, counts, list);
  } else {
    moe_ffn_kernel<<<NEXP * 256, TPB, 0, stream>>>(x, W1, b1, W2, b2,
                                                   out, counts, list);
  }
}
